// MoELayer_39384850104908
// MI455X (gfx1250) — compile-verified, matrix operand order checked by reading, not run on hardware
//
#include <hip/hip_runtime.h>
#include <hip/hip_bf16.h>
#include <cstdint>

typedef __attribute__((ext_vector_type(16))) __bf16 v16bf;
typedef __attribute__((ext_vector_type(8)))  float  v8f;

#define D_MODEL 1024
#define N_EXP   16
#define TOPK    2
#define H_DIM   1024
#define S_DIM   2048
#define NTOK    2048

// ---------------- WMMA helpers ----------------

// 16x32 bf16 fragment from LDS tile stored [rows][32] row-major.
// ISA §7.12.2: lane L holds row (L&15); kbase=(L>>4)*8; e0..7 = K kbase..+7,
// e8..15 = K 16+kbase..+7. Two aligned ds_load_b128 per fragment.
static __device__ __forceinline__ v16bf ld_frag(const __bf16* row, int lane) {
  const int kb = (lane >> 4) << 3;
  union { uint4 q[2]; v16bf v; } u;
  u.q[0] = *reinterpret_cast<const uint4*>(row + kb);
  u.q[1] = *reinterpret_cast<const uint4*>(row + 16 + kb);
  return u.v;
}

static __device__ __forceinline__ v8f wmma_bf16(v16bf a, v16bf b, v8f c) {
  return __builtin_amdgcn_wmma_f32_16x16x32_bf16(false, a, false, b, (short)0, c,
                                                 false, false);
}

static __device__ __forceinline__ float siluf(float x) {
  return x * __builtin_amdgcn_rcpf(1.0f + __expf(-x));
}

// ---------------- Tensor Data Mover ----------------

#if defined(__AMDGCN__) && __has_builtin(__builtin_amdgcn_tensor_load_to_lds)
#define HAVE_TDM 1
typedef __attribute__((ext_vector_type(4))) unsigned int u32x4;
typedef __attribute__((ext_vector_type(8))) int          i32x8;
typedef __attribute__((ext_vector_type(4))) int          i32x4;

static __device__ __forceinline__ void tdm_issue(u32x4 g0, i32x8 g1, i32x4 g2,
                                                 i32x4 g3) {
#if __has_include(<hip/amd_detail/amd_gfx1250_TDM.h>)
  const i32x8 z8 = {0, 0, 0, 0, 0, 0, 0, 0};
  __builtin_amdgcn_tensor_load_to_lds(g0, g1, g2, g3, z8, 0);  // clang-23 arity
#else
  __builtin_amdgcn_tensor_load_to_lds(g0, g1, g2, g3, 0);      // ROCm 7.2 arity
#endif
}

// Dense 2D bf16 tile DMA: rows x cols, row pitch src_stride (elements),
// packed contiguously into LDS (tensor dims == tile dims -> no OOB clip).
static __device__ __forceinline__ void tdm_load_2d_bf16(const __bf16* gsrc,
                                                        void* lds_dst, int rows,
                                                        int cols, int src_stride) {
  const unsigned long long ga = (unsigned long long)(uintptr_t)gsrc;
  u32x4 g0;
  g0[0] = 1u;                                            // count=1, user D#
  g0[1] = (unsigned)(uintptr_t)lds_dst;                  // lds_addr
  g0[2] = (unsigned)ga;
  g0[3] = (unsigned)((ga >> 32) & 0x01FFFFFFull) | (2u << 30);  // addr[56:32]|type=2
  i32x8 g1;
  g1[0] = (1 << 16);                                     // data_size=1 (2B)
  g1[1] = (cols & 0xFFFF) << 16;                         // tensor_dim0 lo
  g1[2] = ((cols >> 16) & 0xFFFF) | ((rows & 0xFFFF) << 16);
  g1[3] = ((rows >> 16) & 0xFFFF) | ((cols & 0xFFFF) << 16);  // | tile_dim0
  g1[4] = (rows & 0xFFFF);                               // tile_dim1
  g1[5] = src_stride;                                    // dim0_stride lo
  g1[6] = 0;
  g1[7] = 0;
  const i32x4 z = {0, 0, 0, 0};
  tdm_issue(g0, g1, z, z);
}

// Gather-mode DMA (§8.7): 16 rows selected by 16-bit indices in groups 2/3;
// each row is 1 x cols at gsrc + idx*src_stride; rows packed into LDS.
static __device__ __forceinline__ void tdm_gather_bf16(const __bf16* gsrc,
                                                       void* lds_dst, int cols,
                                                       int src_stride, i32x4 ilo,
                                                       i32x4 ihi) {
  const unsigned long long ga = (unsigned long long)(uintptr_t)gsrc;
  u32x4 g0;
  g0[0] = 1u | (1u << 31);                               // gather_mode, 16-bit idx
  g0[1] = (unsigned)(uintptr_t)lds_dst;
  g0[2] = (unsigned)ga;
  g0[3] = (unsigned)((ga >> 32) & 0x01FFFFFFull) | (2u << 30);
  i32x8 g1;
  g1[0] = (1 << 16);                                     // data_size=1 (2B)
  g1[1] = (cols & 0xFFFF) << 16;                         // tensor_dim0 = cols
  g1[2] = ((cols >> 16) & 0xFFFF) |
          (((NTOK * TOPK) & 0xFFFF) << 16);              // tensor_dim1 covers rows
  g1[3] = ((cols & 0xFFFF) << 16);                       // tile_dim0 = cols
  g1[4] = 16;                                            // tile_dim1 = #indices
  g1[5] = src_stride;
  g1[6] = 0;
  g1[7] = 0;
  tdm_issue(g0, g1, ilo, ihi);
}

// Pack 16 row indices into group2/3 words, forced into SGPRs via
// readfirstlane so the K-loop's TDM issues carry no per-iteration
// v_readfirstlane/s_mov traffic (indices are loop-invariant).
static __device__ __forceinline__ void pack_gather_idx(
    const int* __restrict__ elist, int e, int cnt, int base, bool toslot,
    i32x4* lo, i32x4* hi) {
  int r[16];
#pragma unroll
  for (int j = 0; j < 16; ++j) {
    const int gr = base + j;
    int v2 = (gr < cnt) ? elist[e * NTOK + gr] : 0;   // clamp -> row 0 (masked later)
    r[j] = toslot ? v2 : (v2 >> 1);
  }
  i32x4 l, h;
#pragma unroll
  for (int w = 0; w < 4; ++w) {
    l[w] = __builtin_amdgcn_readfirstlane((r[2 * w] & 0xFFFF) | (r[2 * w + 1] << 16));
    h[w] = __builtin_amdgcn_readfirstlane((r[8 + 2 * w] & 0xFFFF) |
                                          (r[8 + 2 * w + 1] << 16));
  }
  *lo = l;
  *hi = h;
}
#endif

// ---------------- staged loaders (global fp32 -> regs -> bf16 LDS) ----------

// B tile: 32(K) x 128(N), LDS layout transposed [n][32].
struct BStage { uint4 q0, q1; unsigned n, kq; };

template <bool KMAJOR>
static __device__ __forceinline__ BStage load_b_regs(const float* __restrict__ W,
                                                     int ldb, int k0, int n0,
                                                     int tid) {
  BStage st;
  if (KMAJOR) {              // (k,n) at W[(k0+k)*ldb + n0+n]: 16 coalesced b32
    st.n = tid & 127;
    st.kq = (tid >> 7) << 4;
    const float* s = W + (size_t)(k0 + st.kq) * ldb + (n0 + st.n);
    union { __bf16 h[8]; uint4 q; } u0, u1;
#pragma unroll
    for (int j = 0; j < 8; ++j) u0.h[j] = (__bf16)s[(size_t)j * ldb];
#pragma unroll
    for (int j = 0; j < 8; ++j) u1.h[j] = (__bf16)s[(size_t)(j + 8) * ldb];
    st.q0 = u0.q;
    st.q1 = u1.q;
  } else {                   // (k,n) at W[(n0+n)*ldb + k0+k]: 4x b128
    st.n = tid >> 1;
    st.kq = (tid & 1) << 4;
    const float* s = W + (size_t)(n0 + st.n) * ldb + (k0 + st.kq);
    float4 f0 = *reinterpret_cast<const float4*>(s);
    float4 f1 = *reinterpret_cast<const float4*>(s + 4);
    float4 f2 = *reinterpret_cast<const float4*>(s + 8);
    float4 f3 = *reinterpret_cast<const float4*>(s + 12);
    union { __bf16 h[8]; uint4 q; } u0, u1;
    u0.h[0] = (__bf16)f0.x; u0.h[1] = (__bf16)f0.y; u0.h[2] = (__bf16)f0.z;
    u0.h[3] = (__bf16)f0.w; u0.h[4] = (__bf16)f1.x; u0.h[5] = (__bf16)f1.y;
    u0.h[6] = (__bf16)f1.z; u0.h[7] = (__bf16)f1.w;
    u1.h[0] = (__bf16)f2.x; u1.h[1] = (__bf16)f2.y; u1.h[2] = (__bf16)f2.z;
    u1.h[3] = (__bf16)f2.w; u1.h[4] = (__bf16)f3.x; u1.h[5] = (__bf16)f3.y;
    u1.h[6] = (__bf16)f3.z; u1.h[7] = (__bf16)f3.w;
    st.q0 = u0.q;
    st.q1 = u1.q;
  }
  return st;
}

static __device__ __forceinline__ void store_b_lds(const BStage& st,
                                                   __bf16 (*lb)[32]) {
  *reinterpret_cast<uint4*>(&lb[st.n][st.kq])     = st.q0;
  *reinterpret_cast<uint4*>(&lb[st.n][st.kq + 8]) = st.q1;
}

#ifndef HAVE_TDM
// Fallback A staging when no TDM builtin is available.
struct AStage { uint4 v; unsigned r, k; };

static __device__ __forceinline__ AStage load_a_regs_rows(
    const __bf16* __restrict__ X, int lda, const int* rows, int k0, int tid) {
  AStage st;
  st.r = tid >> 2;
  st.k = (tid & 3) << 3;
  uint4 v = {0u, 0u, 0u, 0u};
  const int row = rows[st.r];
  if (row >= 0)
    v = *reinterpret_cast<const uint4*>(X + (size_t)row * lda + k0 + st.k);
  st.v = v;
  return st;
}

static __device__ __forceinline__ AStage load_a_regs_direct(
    const __bf16* __restrict__ X, int lda, int m0, int k0, int tid) {
  AStage st;
  st.r = tid >> 2;
  st.k = (tid & 3) << 3;
  st.v = *reinterpret_cast<const uint4*>(X + (size_t)(m0 + st.r) * lda + k0 + st.k);
  return st;
}

static __device__ __forceinline__ void store_a_lds(const AStage& st,
                                                   __bf16 (*la)[32]) {
  *reinterpret_cast<uint4*>(&la[st.r][st.k]) = st.v;
}
#endif

// ---------------- kernels ----------------

__global__ void k_convert(const float* __restrict__ x, __bf16* __restrict__ xb,
                          int n, int* __restrict__ counts) {
  const int i = blockIdx.x * blockDim.x + threadIdx.x;
  if (i < n) xb[i] = (__bf16)x[i];
  if (blockIdx.x == 0 && threadIdx.x < N_EXP) counts[threadIdx.x] = 0;
}

__global__ __launch_bounds__(256) void k_router(const float* __restrict__ x,
                                                const float* __restrict__ rw,
                                                int* __restrict__ elist,
                                                float* __restrict__ gbuf,
                                                int* __restrict__ counts) {
  const int t = blockIdx.x;
  const float* xr = x + (size_t)t * D_MODEL;
  __shared__ float part[N_EXP][16];
  __shared__ float logit[N_EXP];
  const int e = threadIdx.x >> 4;
  const int p = threadIdx.x & 15;
  const float* wr = rw + (size_t)e * D_MODEL;
  float s = 0.f;
  for (int d = p; d < D_MODEL; d += 16) s += xr[d] * wr[d];
  part[e][p] = s;
  __syncthreads();
  if (threadIdx.x < N_EXP) {
    float v = 0.f;
    for (int i = 0; i < 16; ++i) v += part[threadIdx.x][i];
    logit[threadIdx.x] = v;
  }
  __syncthreads();
  if (threadIdx.x == 0) {
    int i0 = 0; float b0 = logit[0];
    for (int i = 1; i < N_EXP; ++i)
      if (logit[i] > b0) { b0 = logit[i]; i0 = i; }
    int i1 = (i0 == 0) ? 1 : 0; float b1 = logit[i1];
    for (int i = 0; i < N_EXP; ++i)
      if (i != i0 && logit[i] > b1) { b1 = logit[i]; i1 = i; }
    gbuf[t * 2 + 0] = __builtin_amdgcn_rcpf(1.f + __expf(-b0));
    gbuf[t * 2 + 1] = __builtin_amdgcn_rcpf(1.f + __expf(-b1));
    int p0 = atomicAdd(&counts[i0], 1);
    elist[i0 * NTOK + p0] = t * 2 + 0;
    int p1 = atomicAdd(&counts[i1], 1);
    elist[i1 * NTOK + p1] = t * 2 + 1;
  }
}

// Gathered gate_up GEMM + SiLU*up -> hbuf[slot][H]. 64x128 block tile (x2 for
// gate+up), 8 waves @ 32x32 per matrix. Double-buffered pipeline: wave 0
// issues next A-tile gather TDM + all threads stage next B while computing.
__global__ __launch_bounds__(256) void k_moe_gateup(
    const __bf16* __restrict__ xb, const float* __restrict__ gup,
    const int* __restrict__ elist, const int* __restrict__ counts,
    __bf16* __restrict__ hbuf) {
  const int e = blockIdx.z;
  const int cnt = counts[e];
  const int m0 = blockIdx.y * 64;
  if (m0 >= cnt) return;
  const int n0 = blockIdx.x * 128;
  const float* W = gup + (size_t)e * D_MODEL * (2 * H_DIM);

  __shared__ __align__(16) __bf16 As[2][64][32];
  __shared__ __align__(16) __bf16 Bg[2][128][32];
  __shared__ __align__(16) __bf16 Bu[2][128][32];
  __shared__ int ent_s[64];
#ifndef HAVE_TDM
  __shared__ int row_s[64];
#endif

  const int tid = threadIdx.x;
  if (tid < 64) {
    const int gr = m0 + tid;
    const int ent = (gr < cnt) ? elist[e * NTOK + gr] : -1;
    ent_s[tid] = ent;
#ifndef HAVE_TDM
    row_s[tid] = (ent >= 0) ? (ent >> 1) : -1;
#endif
  }
  __syncthreads();

#ifdef HAVE_TDM
  i32x4 glo[4], ghi[4];
  if (tid < 32) {
#pragma unroll
    for (int d = 0; d < 4; ++d)
      pack_gather_idx(elist, e, cnt, m0 + d * 16, false, &glo[d], &ghi[d]);
  }
#endif

  const int lane = tid & 31;
  const int wid = tid >> 5;
  const int wm = wid >> 2;
  const int wn = wid & 3;
  v8f cg[2][2] = {{{}, {}}, {{}, {}}};
  v8f cu[2][2] = {{{}, {}}, {{}, {}}};

  // ---- prologue: stage k0 = 0 into buffer 0
#ifdef HAVE_TDM
  if (tid < 32) {
#pragma unroll
    for (int d = 0; d < 4; ++d)
      tdm_gather_bf16(xb, &As[0][d * 16][0], 32, D_MODEL, glo[d], ghi[d]);
  }
#else
  AStage as = load_a_regs_rows(xb, D_MODEL, row_s, 0, tid);
#endif
  BStage sg = load_b_regs<true>(W, 2 * H_DIM, 0, n0, tid);
  BStage su = load_b_regs<true>(W, 2 * H_DIM, 0, n0 + H_DIM, tid);
#ifdef HAVE_TDM
  if (tid < 32) __builtin_amdgcn_s_wait_tensorcnt(0);
#else
  store_a_lds(as, As[0]);
#endif
  store_b_lds(sg, Bg[0]);
  store_b_lds(su, Bu[0]);
  __syncthreads();

  int cur = 0;
  for (int k0 = 0; k0 < D_MODEL; k0 += 32) {
    const int nxt = cur ^ 1;
    const bool more = (k0 + 32) < D_MODEL;
    if (more) {
#ifdef HAVE_TDM
      if (tid < 32) {
#pragma unroll
        for (int d = 0; d < 4; ++d)
          tdm_gather_bf16(xb + k0 + 32, &As[nxt][d * 16][0], 32, D_MODEL,
                          glo[d], ghi[d]);
      }
#else
      as = load_a_regs_rows(xb, D_MODEL, row_s, k0 + 32, tid);
#endif
      sg = load_b_regs<true>(W, 2 * H_DIM, k0 + 32, n0, tid);
      su = load_b_regs<true>(W, 2 * H_DIM, k0 + 32, n0 + H_DIM, tid);
    }
    v16bf a0  = ld_frag(&As[cur][wm * 32 + (lane & 15)][0], lane);
    v16bf a1  = ld_frag(&As[cur][wm * 32 + 16 + (lane & 15)][0], lane);
    v16bf bg0 = ld_frag(&Bg[cur][wn * 32 + (lane & 15)][0], lane);
    v16bf bg1 = ld_frag(&Bg[cur][wn * 32 + 16 + (lane & 15)][0], lane);
    v16bf bu0 = ld_frag(&Bu[cur][wn * 32 + (lane & 15)][0], lane);
    v16bf bu1 = ld_frag(&Bu[cur][wn * 32 + 16 + (lane & 15)][0], lane);
    cg[0][0] = wmma_bf16(a0, bg0, cg[0][0]);
    cg[0][1] = wmma_bf16(a0, bg1, cg[0][1]);
    cg[1][0] = wmma_bf16(a1, bg0, cg[1][0]);
    cg[1][1] = wmma_bf16(a1, bg1, cg[1][1]);
    cu[0][0] = wmma_bf16(a0, bu0, cu[0][0]);
    cu[0][1] = wmma_bf16(a0, bu1, cu[0][1]);
    cu[1][0] = wmma_bf16(a1, bu0, cu[1][0]);
    cu[1][1] = wmma_bf16(a1, bu1, cu[1][1]);
    __syncthreads();
    if (more) {
#ifndef HAVE_TDM
      store_a_lds(as, As[nxt]);
#endif
      store_b_lds(sg, Bg[nxt]);
      store_b_lds(su, Bu[nxt]);
#ifdef HAVE_TDM
      if (tid < 32) __builtin_amdgcn_s_wait_tensorcnt(0);
#endif
      __syncthreads();
    }
    cur = nxt;
  }

  const int nl = lane & 15;
  const int rhi = (lane >> 4) * 8;
#pragma unroll
  for (int i = 0; i < 2; ++i) {
    for (int v = 0; v < 8; ++v) {
      const int rloc = wm * 32 + i * 16 + rhi + v;
      const int ent = ent_s[rloc];
      if (ent < 0) continue;
      __bf16* dst = hbuf + (size_t)ent * H_DIM + n0 + wn * 32 + nl;
      dst[0]  = (__bf16)(siluf(cg[i][0][v]) * cu[i][0][v]);
      dst[16] = (__bf16)(siluf(cg[i][1][v]) * cu[i][1][v]);
    }
  }
}

// Gathered down GEMM: obuf[slot][D] = gate * (h[slot] @ down_w[e])
__global__ __launch_bounds__(256) void k_moe_down(
    const __bf16* __restrict__ hbuf, const float* __restrict__ dw,
    const int* __restrict__ elist, const int* __restrict__ counts,
    const float* __restrict__ gbuf, float* __restrict__ obuf) {
  const int e = blockIdx.z;
  const int cnt = counts[e];
  const int m0 = blockIdx.y * 64;
  if (m0 >= cnt) return;
  const int n0 = blockIdx.x * 128;
  const float* W = dw + (size_t)e * H_DIM * D_MODEL;

  __shared__ __align__(16) __bf16 As[2][64][32];
  __shared__ __align__(16) __bf16 Bs[2][128][32];
  __shared__ int ent_s[64];

  const int tid = threadIdx.x;
  if (tid < 64) {
    const int gr = m0 + tid;
    ent_s[tid] = (gr < cnt) ? elist[e * NTOK + gr] : -1;
  }
  __syncthreads();

#ifdef HAVE_TDM
  i32x4 glo[4], ghi[4];
  if (tid < 32) {
#pragma unroll
    for (int d = 0; d < 4; ++d)
      pack_gather_idx(elist, e, cnt, m0 + d * 16, true, &glo[d], &ghi[d]);
  }
#endif

  const int lane = tid & 31;
  const int wid = tid >> 5;
  const int wm = wid >> 2;
  const int wn = wid & 3;
  v8f c[2][2] = {{{}, {}}, {{}, {}}};

#ifdef HAVE_TDM
  if (tid < 32) {
#pragma unroll
    for (int d = 0; d < 4; ++d)
      tdm_gather_bf16(hbuf, &As[0][d * 16][0], 32, H_DIM, glo[d], ghi[d]);
  }
#else
  AStage as = load_a_regs_rows(hbuf, H_DIM, ent_s, 0, tid);
#endif
  BStage sb = load_b_regs<true>(W, D_MODEL, 0, n0, tid);
#ifdef HAVE_TDM
  if (tid < 32) __builtin_amdgcn_s_wait_tensorcnt(0);
#else
  store_a_lds(as, As[0]);
#endif
  store_b_lds(sb, Bs[0]);
  __syncthreads();

  int cur = 0;
  for (int k0 = 0; k0 < H_DIM; k0 += 32) {
    const int nxt = cur ^ 1;
    const bool more = (k0 + 32) < H_DIM;
    if (more) {
#ifdef HAVE_TDM
      if (tid < 32) {
#pragma unroll
        for (int d = 0; d < 4; ++d)
          tdm_gather_bf16(hbuf + k0 + 32, &As[nxt][d * 16][0], 32, H_DIM,
                          glo[d], ghi[d]);
      }
#else
      as = load_a_regs_rows(hbuf, H_DIM, ent_s, k0 + 32, tid);
#endif
      sb = load_b_regs<true>(W, D_MODEL, k0 + 32, n0, tid);
    }
    v16bf a0 = ld_frag(&As[cur][wm * 32 + (lane & 15)][0], lane);
    v16bf a1 = ld_frag(&As[cur][wm * 32 + 16 + (lane & 15)][0], lane);
    v16bf b0 = ld_frag(&Bs[cur][wn * 32 + (lane & 15)][0], lane);
    v16bf b1 = ld_frag(&Bs[cur][wn * 32 + 16 + (lane & 15)][0], lane);
    c[0][0] = wmma_bf16(a0, b0, c[0][0]);
    c[0][1] = wmma_bf16(a0, b1, c[0][1]);
    c[1][0] = wmma_bf16(a1, b0, c[1][0]);
    c[1][1] = wmma_bf16(a1, b1, c[1][1]);
    __syncthreads();
    if (more) {
#ifndef HAVE_TDM
      store_a_lds(as, As[nxt]);
#endif
      store_b_lds(sb, Bs[nxt]);
#ifdef HAVE_TDM
      if (tid < 32) __builtin_amdgcn_s_wait_tensorcnt(0);
#endif
      __syncthreads();
    }
    cur = nxt;
  }

  const int nl = lane & 15;
  const int rhi = (lane >> 4) * 8;
#pragma unroll
  for (int i = 0; i < 2; ++i) {
    for (int v = 0; v < 8; ++v) {
      const int rloc = wm * 32 + i * 16 + rhi + v;
      const int ent = ent_s[rloc];
      if (ent < 0) continue;
      const float g = gbuf[ent];
      float* dst = obuf + (size_t)ent * D_MODEL + n0 + wn * 32 + nl;
      dst[0]  = g * c[i][0][v];
      dst[16] = g * c[i][1][v];
    }
  }
}

// Dense shared-expert gate/up; A-tile via dense TDM, double-buffered.
__global__ __launch_bounds__(256) void k_shared_gateup(
    const __bf16* __restrict__ xb, const float* __restrict__ wg,
    const float* __restrict__ wu, __bf16* __restrict__ hs) {
  const int m0 = blockIdx.y * 64;
  const int n0 = blockIdx.x * 128;

  __shared__ __align__(16) __bf16 As[2][64][32];
  __shared__ __align__(16) __bf16 Bg[2][128][32];
  __shared__ __align__(16) __bf16 Bu[2][128][32];

  const int tid = threadIdx.x;
  const int lane = tid & 31;
  const int wid = tid >> 5;
  const int wm = wid >> 2;
  const int wn = wid & 3;
  v8f cg[2][2] = {{{}, {}}, {{}, {}}};
  v8f cu[2][2] = {{{}, {}}, {{}, {}}};

#ifdef HAVE_TDM
  if (tid < 32)
    tdm_load_2d_bf16(xb + (size_t)m0 * D_MODEL, &As[0][0][0], 64, 32, D_MODEL);
#else
  AStage as = load_a_regs_direct(xb, D_MODEL, m0, 0, tid);
#endif
  BStage sg = load_b_regs<false>(wg, D_MODEL, 0, n0, tid);
  BStage su = load_b_regs<false>(wu, D_MODEL, 0, n0, tid);
#ifdef HAVE_TDM
  if (tid < 32) __builtin_amdgcn_s_wait_tensorcnt(0);
#else
  store_a_lds(as, As[0]);
#endif
  store_b_lds(sg, Bg[0]);
  store_b_lds(su, Bu[0]);
  __syncthreads();

  int cur = 0;
  for (int k0 = 0; k0 < D_MODEL; k0 += 32) {
    const int nxt = cur ^ 1;
    const bool more = (k0 + 32) < D_MODEL;
    if (more) {
#ifdef HAVE_TDM
      if (tid < 32)
        tdm_load_2d_bf16(xb + (size_t)m0 * D_MODEL + k0 + 32, &As[nxt][0][0],
                         64, 32, D_MODEL);
#else
      as = load_a_regs_direct(xb, D_MODEL, m0, k0 + 32, tid);
#endif
      sg = load_b_regs<false>(wg, D_MODEL, k0 + 32, n0, tid);
      su = load_b_regs<false>(wu, D_MODEL, k0 + 32, n0, tid);
    }
    v16bf a0  = ld_frag(&As[cur][wm * 32 + (lane & 15)][0], lane);
    v16bf a1  = ld_frag(&As[cur][wm * 32 + 16 + (lane & 15)][0], lane);
    v16bf bg0 = ld_frag(&Bg[cur][wn * 32 + (lane & 15)][0], lane);
    v16bf bg1 = ld_frag(&Bg[cur][wn * 32 + 16 + (lane & 15)][0], lane);
    v16bf bu0 = ld_frag(&Bu[cur][wn * 32 + (lane & 15)][0], lane);
    v16bf bu1 = ld_frag(&Bu[cur][wn * 32 + 16 + (lane & 15)][0], lane);
    cg[0][0] = wmma_bf16(a0, bg0, cg[0][0]);
    cg[0][1] = wmma_bf16(a0, bg1, cg[0][1]);
    cg[1][0] = wmma_bf16(a1, bg0, cg[1][0]);
    cg[1][1] = wmma_bf16(a1, bg1, cg[1][1]);
    cu[0][0] = wmma_bf16(a0, bu0, cu[0][0]);
    cu[0][1] = wmma_bf16(a0, bu1, cu[0][1]);
    cu[1][0] = wmma_bf16(a1, bu0, cu[1][0]);
    cu[1][1] = wmma_bf16(a1, bu1, cu[1][1]);
    __syncthreads();
    if (more) {
#ifndef HAVE_TDM
      store_a_lds(as, As[nxt]);
#endif
      store_b_lds(sg, Bg[nxt]);
      store_b_lds(su, Bu[nxt]);
#ifdef HAVE_TDM
      if (tid < 32) __builtin_amdgcn_s_wait_tensorcnt(0);
#endif
      __syncthreads();
    }
    cur = nxt;
  }

  const int nl = lane & 15;
  const int rhi = (lane >> 4) * 8;
#pragma unroll
  for (int i = 0; i < 2; ++i) {
    for (int v = 0; v < 8; ++v) {
      const int row = m0 + wm * 32 + i * 16 + rhi + v;
      __bf16* dst = hs + (size_t)row * S_DIM + n0 + wn * 32 + nl;
      dst[0]  = (__bf16)(siluf(cg[i][0][v]) * cu[i][0][v]);
      dst[16] = (__bf16)(siluf(cg[i][1][v]) * cu[i][1][v]);
    }
  }
}

// Shared down-proj fused with final combine:
// out[t][c] = (hs[t] @ Wd^T)[c] + obuf[2t][c] + obuf[2t+1][c]
__global__ __launch_bounds__(256) void k_shared_down(
    const __bf16* __restrict__ hs, const float* __restrict__ wd,
    const float* __restrict__ obuf, float* __restrict__ out) {
  const int m0 = blockIdx.y * 64;
  const int n0 = blockIdx.x * 128;

  __shared__ __align__(16) __bf16 As[2][64][32];
  __shared__ __align__(16) __bf16 Bs[2][128][32];

  const int tid = threadIdx.x;
  const int lane = tid & 31;
  const int wid = tid >> 5;
  const int wm = wid >> 2;
  const int wn = wid & 3;
  v8f c[2][2] = {{{}, {}}, {{}, {}}};

#ifdef HAVE_TDM
  if (tid < 32)
    tdm_load_2d_bf16(hs + (size_t)m0 * S_DIM, &As[0][0][0], 64, 32, S_DIM);
#else
  AStage as = load_a_regs_direct(hs, S_DIM, m0, 0, tid);
#endif
  BStage sb = load_b_regs<false>(wd, S_DIM, 0, n0, tid);
#ifdef HAVE_TDM
  if (tid < 32) __builtin_amdgcn_s_wait_tensorcnt(0);
#else
  store_a_lds(as, As[0]);
#endif
  store_b_lds(sb, Bs[0]);
  __syncthreads();

  int cur = 0;
  for (int k0 = 0; k0 < S_DIM; k0 += 32) {
    const int nxt = cur ^ 1;
    const bool more = (k0 + 32) < S_DIM;
    if (more) {
#ifdef HAVE_TDM
      if (tid < 32)
        tdm_load_2d_bf16(hs + (size_t)m0 * S_DIM + k0 + 32, &As[nxt][0][0],
                         64, 32, S_DIM);
#else
      as = load_a_regs_direct(hs, S_DIM, m0, k0 + 32, tid);
#endif
      sb = load_b_regs<false>(wd, S_DIM, k0 + 32, n0, tid);
    }
    v16bf a0 = ld_frag(&As[cur][wm * 32 + (lane & 15)][0], lane);
    v16bf a1 = ld_frag(&As[cur][wm * 32 + 16 + (lane & 15)][0], lane);
    v16bf b0 = ld_frag(&Bs[cur][wn * 32 + (lane & 15)][0], lane);
    v16bf b1 = ld_frag(&Bs[cur][wn * 32 + 16 + (lane & 15)][0], lane);
    c[0][0] = wmma_bf16(a0, b0, c[0][0]);
    c[0][1] = wmma_bf16(a0, b1, c[0][1]);
    c[1][0] = wmma_bf16(a1, b0, c[1][0]);
    c[1][1] = wmma_bf16(a1, b1, c[1][1]);
    __syncthreads();
    if (more) {
#ifndef HAVE_TDM
      store_a_lds(as, As[nxt]);
#endif
      store_b_lds(sb, Bs[nxt]);
#ifdef HAVE_TDM
      if (tid < 32) __builtin_amdgcn_s_wait_tensorcnt(0);
#endif
      __syncthreads();
    }
    cur = nxt;
  }

  const int nl = lane & 15;
  const int rhi = (lane >> 4) * 8;
#pragma unroll
  for (int i = 0; i < 2; ++i) {
    for (int v = 0; v < 8; ++v) {
      const int row = m0 + wm * 32 + i * 16 + rhi + v;   // token
      const int col = n0 + wn * 32 + nl;
      const float* o0 = obuf + (size_t)(row * 2) * D_MODEL + col;
      const float* o1 = obuf + (size_t)(row * 2 + 1) * D_MODEL + col;
      float* dst = out + (size_t)row * D_MODEL + col;
      dst[0]  = c[i][0][v] + o0[0]  + o1[0];
      dst[16] = c[i][1][v] + o0[16] + o1[16];
    }
  }
}

// ---------------- launcher ----------------

extern "C" void kernel_launch(void* const* d_in, const int* in_sizes, int n_in,
                              void* d_out, int out_size, void* d_ws, size_t ws_size,
                              hipStream_t stream) {
  const float* x   = (const float*)d_in[0];
  const float* rw  = (const float*)d_in[1];
  const float* gup = (const float*)d_in[2];
  const float* dw  = (const float*)d_in[3];
  const float* sg  = (const float*)d_in[4];
  const float* su  = (const float*)d_in[5];
  const float* sd  = (const float*)d_in[6];
  float* out = (float*)d_out;

  char* p = (char*)d_ws;
  auto alloc = [&](size_t bytes) {
    void* r = (void*)p;
    p += (bytes + 255) & ~(size_t)255;
    return r;
  };
  __bf16* xb    = (__bf16*)alloc((size_t)NTOK * D_MODEL * 2);        // 4 MB
  __bf16* hbuf  = (__bf16*)alloc((size_t)NTOK * TOPK * H_DIM * 2);   // 8 MB
  __bf16* hs    = (__bf16*)alloc((size_t)NTOK * S_DIM * 2);          // 8 MB
  float*  obuf  = (float*)alloc((size_t)NTOK * TOPK * D_MODEL * 4);  // 16 MB
  int*    elist = (int*)alloc((size_t)N_EXP * NTOK * 4);             // 128 KB
  float*  gbuf  = (float*)alloc((size_t)NTOK * TOPK * 4);            // 16 KB
  int*    counts = (int*)alloc(256);

  k_convert<<<(NTOK * D_MODEL) / 256, 256, 0, stream>>>(x, xb, NTOK * D_MODEL, counts);
  k_router<<<NTOK, 256, 0, stream>>>(x, rw, elist, gbuf, counts);
  k_moe_gateup<<<dim3(H_DIM / 128, NTOK / 64, N_EXP), 256, 0, stream>>>(
      xb, gup, elist, counts, hbuf);
  k_moe_down<<<dim3(D_MODEL / 128, NTOK / 64, N_EXP), 256, 0, stream>>>(
      hbuf, dw, elist, counts, gbuf, obuf);
  k_shared_gateup<<<dim3(S_DIM / 128, NTOK / 64), 256, 0, stream>>>(xb, sg, su, hs);
  k_shared_down<<<dim3(D_MODEL / 128, NTOK / 64), 256, 0, stream>>>(hs, sd, obuf, out);
}
